// RPS_XTM_82635170775708
// MI455X (gfx1250) — matrix operand order repaired and checked by reading, not compiled, not run on hardware
//
#include <hip/hip_runtime.h>
#include <stddef.h>

// Problem constants (from the reference)
#define Bsz   2048
#define Vsz   30000
#define Ksz   200
#define ENsz  200
#define Csz   50
#define NPAD  208      // EN padded to 13*16 tiles for encoder GEMM
#define KPAD  224      // K padded to 7*32 for decoder GEMM reduction
#define VPADK 30016    // V padded to 938*32 for encoder GEMM reduction
#define KCH1  938      // VPADK/32 : encoder K chunks
#define KCH2  7        // KPAD/32  : decoder K chunks
#define MT    128      // Bsz/16   : M tiles
#define NT1   13       // NPAD/16  : encoder N tiles
#define NT2   1875     // Vsz/16   : decoder N tiles
#define TILEE 512      // elements per 32x16 bf16 fragment tile
#define SPLITK 4       // encoder GEMM K-split factor
#define SEGCH  235     // ceil(KCH1 / SPLITK)
#define LOGV2 (-0.00501254182f)  // log(1 - 1/200)
#define VAR2  0.995f

typedef __attribute__((ext_vector_type(16))) __bf16 v16bf;
typedef __attribute__((ext_vector_type(8)))  float  v8f;

// Fragment layout (CDNA5 ISA 7.12.2, wave32, 16-bit operands):
//  A 16x32 (MxK): lane l holds row m=l&15; element j -> k = (j<8)? (l>>4)*8+j
//                 : 16+(l>>4)*8+(j-8).   Inverse: half=(k>>3)&1, j=(k&7)+8*(k>>4).
//  B 32x16 (KxN): lane l holds col n=l&15; element j -> k = (j<8)? (l>>4)*8+j : 16+(l>>4)*8+(j-8).
// Tiles are stored pre-swizzled as 512 contiguous bf16 = [lane][j], so a
// fragment is one aligned 32-byte vector load per lane.

// ---------------------------------------------------------------- utilities

__global__ void zero_kernel(float* p, int n) {
  int i = blockIdx.x * blockDim.x + threadIdx.x;
  if (i < n) p[i] = 0.0f;
}

__device__ __forceinline__ float blk_reduce_sum(float v, float* red) {
  int tid = threadIdx.x;
  red[tid] = v; __syncthreads();
  for (int s = blockDim.x >> 1; s > 0; s >>= 1) {
    if (tid < s) red[tid] += red[tid + s];
    __syncthreads();
  }
  float r = red[0]; __syncthreads();
  return r;
}

__device__ __forceinline__ float blk_reduce_max(float v, float* red) {
  int tid = threadIdx.x;
  red[tid] = v; __syncthreads();
  for (int s = blockDim.x >> 1; s > 0; s >>= 1) {
    if (tid < s) red[tid] = fmaxf(red[tid], red[tid + s]);
    __syncthreads();
  }
  float r = red[0]; __syncthreads();
  return r;
}

__device__ __forceinline__ float softplusf(float x) {
  return (x > 20.0f) ? x : log1pf(expf(x));
}

// ------------------------------------------------- weight packing (swizzled)

// W1 [30000][200] f32 -> bf16 fragment tiles: tile id = ntile*KCH1 + ch,
// element e = lane*16 + j;  k = ch*32 + (j>>3)*16 + (lane>>4)*8 + (j&7),  n = ntile*16 + (lane&15)
__global__ void pack_w1_kernel(const float* __restrict__ w1, __bf16* __restrict__ w1p) {
  int idx = blockIdx.x * blockDim.x + threadIdx.x;
  if (idx >= NT1 * KCH1 * TILEE) return;
  int tile = idx >> 9, e = idx & 511;
  int lane = e >> 4, j = e & 15;
  int ntile = tile / KCH1, ch = tile % KCH1;
  int k = ch * 32 + ((j >> 3) << 4) + ((lane >> 4) << 3) + (j & 7);
  int n = ntile * 16 + (lane & 15);
  float v = (k < Vsz && n < ENsz) ? w1[(size_t)k * ENsz + n] : 0.0f;
  w1p[idx] = (__bf16)v;
}

// phi [200][30000] f32 -> bf16 fragment tiles: tile id = ntile*KCH2 + ch
__global__ void pack_phi_kernel(const float* __restrict__ phi, __bf16* __restrict__ phip) {
  int idx = blockIdx.x * blockDim.x + threadIdx.x;
  if (idx >= NT2 * KCH2 * TILEE) return;
  int tile = idx >> 9, e = idx & 511;
  int lane = e >> 4, j = e & 15;
  int ntile = tile / KCH2, ch = tile % KCH2;
  int k = ch * 32 + ((j >> 3) << 4) + ((lane >> 4) << 3) + (j & 7);
  int n = ntile * 16 + (lane & 15);
  float v = (k < Ksz) ? phi[(size_t)k * Vsz + n] : 0.0f;
  phip[idx] = (__bf16)v;
}

// --------------------------------------------- encoder GEMM1 (WMMA bf16)
// H1acc += X @ W1 (split-K partials via f32 atomics). One wave per 16x16
// output tile per K-segment. A staged via LDS in fragment order with
// guard-free packed-pair stores except for the single partial tail chunk;
// B read directly from pre-swizzled w1p (32B vector per lane).

template <bool GUARD>
__device__ __forceinline__ void stage_a(const float* __restrict__ xm, int kc,
                                        __bf16* __restrict__ af, int lane) {
#pragma unroll
  for (int t = 0; t < 8; ++t) {
    int pidx = t * 64 + lane * 2;       // pair-element index, k even
    int r = pidx >> 5;                  // row in tile
    int k = pidx & 31;
    int gk = kc + k;
    float lo = 0.0f, hi = 0.0f;
    if (!GUARD || gk < Vsz) {           // gk even, Vsz even -> gk+1 also valid
      const float* xp = xm + (size_t)r * Vsz + gk;
      lo = xp[0]; hi = xp[1];
    }
    int h = (k >> 3) & 1;
    int j = (k & 7) + ((k >> 4) << 3);  // even since k even
    int dl = r + (h << 4);
    union { __bf16 hh[2]; unsigned u; } pk;
    pk.hh[0] = (__bf16)lo; pk.hh[1] = (__bf16)hi;
    *(unsigned*)&af[dl * 16 + j] = pk.u;
  }
}

__global__ void __launch_bounds__(32) gemm1_kernel(const float* __restrict__ x,
                                                   const __bf16* __restrict__ w1p,
                                                   float* __restrict__ h1acc) {
  __shared__ v16bf lsAf[32];
  __bf16* af = (__bf16*)lsAf;
  const int lane = threadIdx.x;
  const int m0 = blockIdx.x * 16;
  const int half = lane >> 4;
  const float* xm = x + (size_t)m0 * Vsz;
  const __bf16* Bp = w1p + ((size_t)blockIdx.y * KCH1) * TILEE + lane * 16;
  const int ch0 = blockIdx.z * SEGCH;
  const int ch1 = (ch0 + SEGCH < KCH1) ? ch0 + SEGCH : KCH1;
  const int chF = (ch1 < KCH1 - 1) ? ch1 : KCH1 - 1;  // last chunk is partial
  v8f acc = {};
  // guard-free main loop
  for (int ch = ch0; ch < chF; ++ch) {
    __syncthreads();
    stage_a<false>(xm, ch * 32, af, lane);
    {  // clamped prefetch of next chunk (global_prefetch_b8), no branch
      int pk2 = ch * 32 + 32 + (half << 4);
      pk2 = (pk2 < Vsz - 1) ? pk2 : (Vsz - 1);
      __builtin_prefetch(&xm[(size_t)(lane & 15) * Vsz + pk2], 0, 1);
    }
    __syncthreads();
    v16bf a = lsAf[lane];
    v16bf b = *(const v16bf*)(Bp + (size_t)ch * TILEE);
    acc = __builtin_amdgcn_wmma_f32_16x16x32_bf16(false, a, false, b, (short)0, acc,
                                                  false, false);
  }
  // guarded tail chunk (only the segment owning chunk KCH1-1 runs this)
  for (int ch = (ch0 > KCH1 - 1 ? ch0 : KCH1 - 1); ch < ch1; ++ch) {
    __syncthreads();
    stage_a<true>(xm, ch * 32, af, lane);
    __syncthreads();
    v16bf a = lsAf[lane];
    v16bf b = *(const v16bf*)(Bp + (size_t)ch * TILEE);
    acc = __builtin_amdgcn_wmma_f32_16x16x32_bf16(false, a, false, b, (short)0, acc,
                                                  false, false);
  }
  // accumulate split-K partials (h1acc pre-zeroed, padded to NPAD columns)
  int nn = blockIdx.y * 16 + (lane & 15);
#pragma unroll
  for (int r = 0; r < 8; ++r) {
    atomicAdd(&h1acc[(size_t)(m0 + half * 8 + r) * NPAD + nn], acc[r]);
  }
}

// bias + softplus epilogue for the encoder GEMM
__global__ void h1_epilogue_kernel(const float* __restrict__ h1acc,
                                   const float* __restrict__ b1, float* __restrict__ h1) {
  int idx = blockIdx.x * blockDim.x + threadIdx.x;
  if (idx >= Bsz * ENsz) return;
  int i = idx / ENsz, j = idx % ENsz;
  h1[idx] = softplusf(h1acc[(size_t)i * NPAD + j] + b1[j]);
}

// ---------------------------------------------- small FC layers (scalar f32)

__global__ void fc2_kernel(const float* __restrict__ h1, const float* __restrict__ w2,
                           const float* __restrict__ b2, float* __restrict__ h2) {
  int idx = blockIdx.x * blockDim.x + threadIdx.x;
  if (idx >= Bsz * ENsz) return;
  int i = idx / ENsz, j = idx % ENsz;
  const float* hr = h1 + (size_t)i * ENsz;
  float s = b2[j];
  for (int t = 0; t < ENsz; ++t) s += hr[t] * w2[t * ENsz + j];
  h2[idx] = softplusf(s);
}

__global__ void heads_kernel(const float* __restrict__ h2, const float* __restrict__ wmu,
                             const float* __restrict__ bmu, const float* __restrict__ wlv,
                             const float* __restrict__ blv, const float* __restrict__ eps,
                             float* __restrict__ mu, float* __restrict__ lv,
                             float* __restrict__ z) {
  int idx = blockIdx.x * blockDim.x + threadIdx.x;
  if (idx >= Bsz * Ksz) return;
  int i = idx / Ksz, j = idx % Ksz;
  const float* hr = h2 + (size_t)i * ENsz;
  float sm_ = bmu[j], sl = blv[j];
  for (int t = 0; t < ENsz; ++t) {
    float h = hr[t];
    sm_ += h * wmu[t * Ksz + j];
    sl += h * wlv[t * Ksz + j];
  }
  mu[idx] = sm_;
  lv[idx] = sl;
  z[idx] = sm_ + expf(0.5f * sl) * eps[idx];
}

// ------ per-row softmax(theta) written in decoder-A fragment order + KLD term

__global__ void theta_kld_kernel(const float* __restrict__ z, const float* __restrict__ mu,
                                 const float* __restrict__ lv, __bf16* __restrict__ theta_t,
                                 float* __restrict__ kld_acc) {
  __shared__ float red[256];
  int i = blockIdx.x, tid = threadIdx.x;
  const float* zr = z + (size_t)i * Ksz;
  float val = (tid < Ksz) ? zr[tid] : -3.4e38f;
  float zmax = blk_reduce_max(val, red);
  float e = (tid < Ksz) ? expf(zr[tid] - zmax) : 0.0f;
  float s = blk_reduce_sum(e, red);
  if (tid < KPAD) {
    // scatter into fragment-tile order: tile = (i/16)*KCH2 + tid/32
    int mtile = i >> 4, m = i & 15;
    int ch = tid >> 5, kk = tid & 31;
    int h = (kk >> 3) & 1;
    int j = (kk & 7) + ((kk >> 4) << 3);
    int lane = m + (h << 4);
    size_t pos = ((size_t)(mtile * KCH2 + ch) << 9) + lane * 16 + j;
    theta_t[pos] = (__bf16)((tid < Ksz) ? (e / s) : 0.0f);
  }
  float kt = 0.0f;
  if (tid < Ksz) {
    float m = mu[(size_t)i * Ksz + tid];
    float l = lv[(size_t)i * Ksz + tid];
    kt = expf(l) / VAR2 + m * m / VAR2 + LOGV2 - l;
  }
  float ks = blk_reduce_sum(kt, red);
  if (tid == 0) atomicAdd(kld_acc, 0.5f * (ks - (float)Ksz));
}

// ------------- decoder GEMM (WMMA bf16, LDS-free: both operands pre-swizzled)
// + fused column sum / sum-of-squares for BatchNorm statistics.

__global__ void __launch_bounds__(32) gemm2_kernel(const __bf16* __restrict__ theta_t,
                                                   const __bf16* __restrict__ phip,
                                                   float* __restrict__ Y,
                                                   float* __restrict__ colsum,
                                                   float* __restrict__ colsumsq) {
  const int lane = threadIdx.x;
  const int mt = blockIdx.x, nt = blockIdx.y;
  const int half = lane >> 4;
  const __bf16* Ap = theta_t + ((size_t)mt * KCH2) * TILEE + lane * 16;
  const __bf16* Bp = phip + ((size_t)nt * KCH2) * TILEE + lane * 16;
  v8f acc = {};
#pragma unroll
  for (int ch = 0; ch < KCH2; ++ch) {
    v16bf a = *(const v16bf*)(Ap + (size_t)ch * TILEE);
    v16bf b = *(const v16bf*)(Bp + (size_t)ch * TILEE);
    acc = __builtin_amdgcn_wmma_f32_16x16x32_bf16(false, a, false, b, (short)0, acc,
                                                  false, false);
  }
  int n = lane & 15;
  float ps = 0.0f, pss = 0.0f;
#pragma unroll
  for (int r = 0; r < 8; ++r) {
    float v = acc[r];
    Y[(size_t)(mt * 16 + half * 8 + r) * Vsz + nt * 16 + n] = v;
    ps += v;
    pss += v * v;
  }
  // lanes l and l^16 hold the other 8 rows of the same column
  ps += __shfl_xor(ps, 16, 32);
  pss += __shfl_xor(pss, 16, 32);
  if (lane < 16) {
    atomicAdd(&colsum[nt * 16 + n], ps);
    atomicAdd(&colsumsq[nt * 16 + n], pss);
  }
}

__global__ void stats_final_kernel(const float* __restrict__ colsum,
                                   const float* __restrict__ colsumsq,
                                   float* __restrict__ colmean, float* __restrict__ colrstd) {
  int i = blockIdx.x * blockDim.x + threadIdx.x;
  if (i >= Vsz) return;
  float mean = colsum[i] * (1.0f / (float)Bsz);
  float var = colsumsq[i] * (1.0f / (float)Bsz) - mean * mean;
  colmean[i] = mean;
  colrstd[i] = rsqrtf(var + 1e-5f);
}

// ---------------- per-row BN -> softmax -> reconstruction loss (row in LDS)

__global__ void rowpass_kernel(const float* __restrict__ Y, const float* __restrict__ xbow,
                               const float* __restrict__ colmean,
                               const float* __restrict__ colrstd,
                               const float* __restrict__ bias, float* __restrict__ rec_acc) {
  extern __shared__ float sm[];  // [blockDim.x red][Vsz row]
  float* red = sm;
  float* row = sm + blockDim.x;
  int i = blockIdx.x, tid = threadIdx.x;
  const float* y = Y + (size_t)i * Vsz;
  float lmax = -3.4e38f;
  for (int v = tid; v < Vsz; v += blockDim.x) {
    float t = (y[v] - colmean[v]) * colrstd[v] + bias[v];
    row[v] = t;
    lmax = fmaxf(lmax, t);
  }
  float rmax = blk_reduce_max(lmax, red);
  float lsum = 0.0f;
  for (int v = tid; v < Vsz; v += blockDim.x) lsum += expf(row[v] - rmax);
  float sumexp = blk_reduce_sum(lsum, red);
  float inv = 1.0f / sumexp;
  const float* xb = xbow + (size_t)i * Vsz;
  float lloss = 0.0f;
  for (int v = tid; v < Vsz; v += blockDim.x) {
    float p = expf(row[v] - rmax) * inv;
    lloss += xb[v] * logf(p + 1e-10f);
  }
  float tot = blk_reduce_sum(lloss, red);
  if (tid == 0) atomicAdd(rec_acc, -tot);
}

// ------------------------------------------------------------ DCL loss

__global__ void __launch_bounds__(256) dcl_kernel(const float* __restrict__ za,
                                                  const float* __restrict__ zb,
                                                  const int* __restrict__ la,
                                                  const int* __restrict__ lb,
                                                  float* __restrict__ dcl_acc) {
  __shared__ float pA[Csz * Ksz];
  __shared__ float pB[Csz * Ksz];
  __shared__ float lg[Csz * Csz];
  __shared__ float cA[Csz], cB[Csz], nrmA[Csz], nrmB[Csz], pres[Csz];
  __shared__ float red[256];
  int tid = threadIdx.x;
  for (int p = tid; p < Csz * Ksz; p += 256) { pA[p] = 0.0f; pB[p] = 0.0f; }
  if (tid < Csz) { cA[tid] = 0.0f; cB[tid] = 0.0f; }
  __syncthreads();
  for (int i = tid; i < Bsz; i += 256) {
    atomicAdd(&cA[la[i]], 1.0f);
    atomicAdd(&cB[lb[i]], 1.0f);
  }
  for (int p = tid; p < Bsz * Ksz; p += 256) {
    int i = p / Ksz, k = p % Ksz;
    atomicAdd(&pA[la[i] * Ksz + k], za[p]);
    atomicAdd(&pB[lb[i] * Ksz + k], zb[p]);
  }
  __syncthreads();
  for (int p = tid; p < Csz * Ksz; p += 256) {
    int c = p / Ksz;
    pA[p] /= (cA[c] + 1e-8f);
    pB[p] /= (cB[c] + 1e-8f);
  }
  __syncthreads();
  if (tid < Csz) {
    float sa = 0.0f, sb = 0.0f;
    for (int k = 0; k < Ksz; ++k) {
      float a = pA[tid * Ksz + k], b = pB[tid * Ksz + k];
      sa += a * a; sb += b * b;
    }
    nrmA[tid] = sqrtf(sa);
    nrmB[tid] = sqrtf(sb);
    pres[tid] = (cA[tid] > 0.0f || cB[tid] > 0.0f) ? 1.0f : 0.0f;
  }
  __syncthreads();
  for (int p = tid; p < Csz * Csz; p += 256) {
    int i = p / Csz, j = p % Csz;
    float d = 0.0f;
    for (int k = 0; k < Ksz; ++k) d += pA[i * Ksz + k] * pB[j * Ksz + k];
    lg[p] = (d / 0.07f) / (nrmA[i] + 1e-8f) / (nrmB[j] + 1e-8f);
  }
  __syncthreads();
  float lossc = 0.0f, presc = 0.0f;
  if (tid < Csz) {
    float den = 0.0f;
    for (int j = 0; j < Csz; ++j) {
      float c = fminf(fmaxf(lg[tid * Csz + j], -1e4f), 1e4f);
      den += expf(c) * pres[j];
    }
    float num = expf(lg[tid * Csz + tid]);
    float li = -logf((num + 1e-8f) / (den + 1e-8f));
    lossc = li * pres[tid];
    presc = pres[tid];
  }
  float sl = blk_reduce_sum(lossc, red);
  float sp = blk_reduce_sum(presc, red);
  if (tid == 0) atomicAdd(dcl_acc, sl / sp);
}

// ------------------------------------------------------------ final output

__global__ void final_kernel(const float* __restrict__ sc, float* __restrict__ out) {
  // sc: [0]=rec_en [1]=kld_en [2]=rec_cn [3]=kld_cn [4]=dcl
  float tm = (sc[0] + sc[1]) / (float)Bsz + (sc[2] + sc[3]) / (float)Bsz;
  float dcl = sc[4];
  out[0] = tm;
  out[1] = dcl;
  out[2] = tm + dcl;
}

// ------------------------------------------------------------ host launcher

extern "C" void kernel_launch(void* const* d_in, const int* in_sizes, int n_in,
                              void* d_out, int out_size, void* d_ws, size_t ws_size,
                              hipStream_t stream) {
  const float* x[2]    = {(const float*)d_in[0], (const float*)d_in[1]};
  const float* xbow[2] = {(const float*)d_in[2], (const float*)d_in[3]};
  const int* lab_en  = (const int*)d_in[4];
  const int* lab_cn  = (const int*)d_in[5];
  const int* lab_c2e = (const int*)d_in[6];
  const int* lab_e2c = (const int*)d_in[7];
  const float* eps[2] = {(const float*)d_in[8], (const float*)d_in[9]};
  const float* W1 = (const float*)d_in[10];
  const float* b1 = (const float*)d_in[11];
  const float* W2 = (const float*)d_in[12];
  const float* b2 = (const float*)d_in[13];
  const float* Wmu = (const float*)d_in[14];
  const float* bmu = (const float*)d_in[15];
  const float* Wlv = (const float*)d_in[16];
  const float* blv = (const float*)d_in[17];
  const float* phi[2]  = {(const float*)d_in[18], (const float*)d_in[19]};
  const float* bias[2] = {(const float*)d_in[20], (const float*)d_in[21]};
  float* out = (float*)d_out;

  const size_t theta_tiles = (size_t)MT * KCH2 * TILEE;  // per-language theta tile buf

  // workspace carve-up (~294 MB total)
  char* ws = (char*)d_ws;
  size_t off = 0;
  auto carve = [&](size_t bytes) -> char* {
    char* p = ws + off;
    off = (off + bytes + 255) & ~(size_t)255;
    return p;
  };
  float*  sc       = (float*)carve(16 * sizeof(float));                 // accumulators
  __bf16* w1p      = (__bf16*)carve((size_t)NT1 * KCH1 * TILEE * 2);
  float*  h1acc    = (float*)carve((size_t)Bsz * NPAD * 4);             // reused per lang
  float*  h1       = (float*)carve((size_t)2 * Bsz * ENsz * 4);
  float*  h2       = (float*)carve((size_t)2 * Bsz * ENsz * 4);
  float*  mu       = (float*)carve((size_t)2 * Bsz * Ksz * 4);
  float*  lv       = (float*)carve((size_t)2 * Bsz * Ksz * 4);
  float*  z        = (float*)carve((size_t)2 * Bsz * Ksz * 4);
  __bf16* theta_t  = (__bf16*)carve(2 * theta_tiles * 2);
  __bf16* phip     = (__bf16*)carve((size_t)NT2 * KCH2 * TILEE * 2);    // reused per lang
  float*  colstats = (float*)carve((size_t)2 * Vsz * 4);                // sum | sumsq
  float*  colmean  = (float*)carve((size_t)Vsz * 4);
  float*  colrstd  = (float*)carve((size_t)Vsz * 4);
  float*  Y        = (float*)carve((size_t)Bsz * Vsz * 4);              // reused per lang
  float* colsum = colstats;
  float* colsumsq = colstats + Vsz;

  // 0) zero accumulators
  zero_kernel<<<1, 64, 0, stream>>>(sc, 16);

  // 1) pack W1 -> swizzled bf16 fragment tiles
  {
    int n = NT1 * KCH1 * TILEE;
    pack_w1_kernel<<<(n + 255) / 256, 256, 0, stream>>>(W1, w1p);
  }

  // 2) encoder GEMM1 (split-K WMMA) + bias/softplus epilogue, both languages
  for (int lg = 0; lg < 2; ++lg) {
    zero_kernel<<<(Bsz * NPAD + 255) / 256, 256, 0, stream>>>(h1acc, Bsz * NPAD);
    gemm1_kernel<<<dim3(MT, NT1, SPLITK), 32, 0, stream>>>(x[lg], w1p, h1acc);
    h1_epilogue_kernel<<<(Bsz * ENsz + 255) / 256, 256, 0, stream>>>(
        h1acc, b1, h1 + (size_t)lg * Bsz * ENsz);
  }

  // 3) small FC layers + reparameterization + theta/KLD
  for (int lg = 0; lg < 2; ++lg) {
    int n = Bsz * ENsz;
    fc2_kernel<<<(n + 255) / 256, 256, 0, stream>>>(
        h1 + (size_t)lg * Bsz * ENsz, W2, b2, h2 + (size_t)lg * Bsz * ENsz);
    heads_kernel<<<(Bsz * Ksz + 255) / 256, 256, 0, stream>>>(
        h2 + (size_t)lg * Bsz * ENsz, Wmu, bmu, Wlv, blv, eps[lg],
        mu + (size_t)lg * Bsz * Ksz, lv + (size_t)lg * Bsz * Ksz,
        z + (size_t)lg * Bsz * Ksz);
    theta_kld_kernel<<<Bsz, 256, 0, stream>>>(
        z + (size_t)lg * Bsz * Ksz, mu + (size_t)lg * Bsz * Ksz,
        lv + (size_t)lg * Bsz * Ksz, theta_t + lg * theta_tiles, &sc[1 + 2 * lg]);
  }

  // 4) decoder GEMM + BN + softmax + reconstruction loss, per language
  for (int lg = 0; lg < 2; ++lg) {
    int n = NT2 * KCH2 * TILEE;
    pack_phi_kernel<<<(n + 255) / 256, 256, 0, stream>>>(phi[lg], phip);
    zero_kernel<<<(2 * Vsz + 255) / 256, 256, 0, stream>>>(colstats, 2 * Vsz);
    gemm2_kernel<<<dim3(MT, NT2), 32, 0, stream>>>(
        theta_t + lg * theta_tiles, phip, Y, colsum, colsumsq);
    stats_final_kernel<<<(Vsz + 255) / 256, 256, 0, stream>>>(colsum, colsumsq,
                                                              colmean, colrstd);
    size_t shmem = (size_t)(Vsz + 256) * sizeof(float);  // 120 KB row fits in 320 KB LDS
    rowpass_kernel<<<Bsz, 256, shmem, stream>>>(Y, xbow[lg], colmean, colrstd,
                                                bias[lg], &sc[0 + 2 * lg]);
  }

  // 5) DCL loss (both directions), accumulate into sc[4]
  dcl_kernel<<<1, 256, 0, stream>>>(z, z + (size_t)Bsz * Ksz, lab_en, lab_e2c, &sc[4]);
  dcl_kernel<<<1, 256, 0, stream>>>(z + (size_t)Bsz * Ksz, z, lab_cn, lab_c2e, &sc[4]);

  // 6) final scalar assembly: (tm, dcl, tm+dcl)
  final_kernel<<<1, 1, 0, stream>>>(sc, out);
}
